// Node_edge_embedding_88656714925200
// MI455X (gfx1250) — matrix operand order repaired and checked by reading, not compiled, not run on hardware
//
#include <hip/hip_runtime.h>
#include <hip/hip_bf16.h>

// ---------------- problem constants (fixed by the reference) ----------------
#define NN      25000      // nodes
#define EE      400000     // edges
#define R_DIM   64
#define U_DIM   64
#define DIM     128        // edge-MLP input
#define H_E     512        // edge-MLP hidden
#define OUT_E   128        // edge-MLP output (= 16 reps * 4 L * 2)
#define H_N     256        // node-MLP hidden
#define N_SC    64         // node scalars
#define TM      64         // edges per workgroup (4 M-blocks of 16)

typedef __attribute__((ext_vector_type(16))) _Float16 v16h;
typedef __attribute__((ext_vector_type(8)))  float    v8f;

// ------------------------------ WMMA helpers -------------------------------
// A fragment (16x32 f16, M rows from LDS row-major [M][Kstride]):
//   lanes 0-15 : row = m0+lane,    halves 0-7 = K[k0+0..7],  halves 8-15 = K[k0+16..23]
//   lanes 16-31: row = m0+lane-16, halves 0-7 = K[k0+8..15], halves 8-15 = K[k0+24..31]
__device__ __forceinline__ v16h load_a_lds(const _Float16* base, int stride_h, int m0, int k0) {
    int lane = threadIdx.x & 31;
    int r  = m0 + (lane & 15);
    int kb = k0 + ((lane < 16) ? 0 : 8);
    const _Float16* p = base + (size_t)r * stride_h + kb;
    v16h out;
    ((uint4*)&out)[0] = *(const uint4*)(p);        // 8 halves
    ((uint4*)&out)[1] = *(const uint4*)(p + 16);   // 8 halves, +16 halves ahead
    return out;
}

// B fragment (32x16 f16) from pre-packed weights: tile = 32 lanes x 16 halves
// contiguous (1KB). Tiles ordered k-inner: off = (tn*tiles_k + tk)*512 halves.
__device__ __forceinline__ v16h load_b(const _Float16* w, int tiles_k, int tk, int tn) {
    int lane = threadIdx.x & 31;
    const _Float16* p = w + ((size_t)(tn * tiles_k + tk) * 512) + lane * 16;
    v16h out;
    ((uint4*)&out)[0] = *(const uint4*)(p);
    ((uint4*)&out)[1] = *(const uint4*)(p + 8);
    return out;
}

__device__ __forceinline__ v8f wmma_f16(v16h a, v16h b, v8f c) {
    return __builtin_amdgcn_wmma_f32_16x16x32_f16(false, a, false, b, (short)0, c, false, false);
}

// D (16x16 f32): VGPR i -> M = i + (lane<16 ? 0 : 8), N = n0 + (lane&15)
__device__ __forceinline__ void store_d_relu(_Float16* dst, int stride_h, int m0, int n0,
                                             v8f acc, const float* bias) {
    int lane = threadIdx.x & 31;
    int n  = n0 + (lane & 15);
    int mb = m0 + ((lane < 16) ? 0 : 8);
    float bv = bias[n];
#pragma unroll
    for (int i = 0; i < 8; ++i) {
        float v = acc[i] + bv;
        v = v > 0.f ? v : 0.f;
        dst[(size_t)(mb + i) * stride_h + n] = (_Float16)v;
    }
}

__device__ __forceinline__ void atomicMaxF(float* p, float v) {
    if (v >= 0.f) atomicMax((int*)p, __float_as_int(v));
    else          atomicMin((unsigned int*)p, __float_as_uint(v));
}

// ------------------------- weight pre-pack (f32->f16) ----------------------
// dst layout: tiles of 16x32 (N x K) B-fragments, 512 halves/tile, k-inner.
__global__ void prep_weight(const float* __restrict__ src, _Float16* __restrict__ dst,
                            int K, int N) {
    int idx = blockIdx.x * 256 + threadIdx.x;
    if (idx >= K * N) return;
    int t    = idx >> 9;
    int rem  = idx & 511;
    int lane = rem >> 4;
    int h    = rem & 15;
    int tiles_k = K >> 5;
    int tn = t / tiles_k;
    int tk = t - tn * tiles_k;
    int kk = tk * 32 + ((h < 8) ? h : 8 + h) + ((lane < 16) ? 0 : 8);
    int nn = tn * 16 + (lane & 15);
    dst[idx] = (_Float16)src[(size_t)kk * N + nn];
}

// ------------------------------- init buffers ------------------------------
__global__ void init_kernel(float* nodeMax, float* nodeSum, float* out_edge) {
    int i = blockIdx.x * 256 + threadIdx.x;
    if (i < NN * OUT_E) out_edge[i] = 0.f;
    if (i < NN) { nodeMax[i] = -__builtin_inff(); nodeSum[i] = 0.f; }
}

// ------------------------ fused edge MLP + LN + alpha ----------------------
// One block = 64 edges (4 M-blocks). 8 waves each own a 64-col N-slice and all
// 4 M-blocks, so every B fragment fetched from L2 feeds 4 WMMAs. N-work is
// done in two half-sweeps (2 N-tiles live) to cap accumulator pressure at
// 2x4x8 = 64 VGPRs; each B tile is still loaded exactly once per WG.
__global__ __launch_bounds__(256) void edge_mlp(
    const float* __restrict__ dist, const float* __restrict__ u, const int* __restrict__ ei,
    const _Float16* __restrict__ W1, const float* __restrict__ b1,
    const _Float16* __restrict__ W2, const float* __restrict__ b2,
    const _Float16* __restrict__ W3, const float* __restrict__ b3,
    const float* __restrict__ ln_g, const float* __restrict__ ln_b,
    const float* __restrict__ aW,   const float* __restrict__ ab,
    _Float16* __restrict__ embH, float* __restrict__ logit, float* __restrict__ nodeMax)
{
    // carved LDS: Xs [0,16K), H1s [16K,80K), H2s [80K,144K); Es (32K f32)
    // overlays Xs+H1s (both dead once layer 3 starts).
    __shared__ __align__(16) char smem[(16 + 64 + 64) * 1024];
    _Float16* Xs  = (_Float16*)smem;                   // 64 x 128 f16
    _Float16* H1s = (_Float16*)(smem + 16 * 1024);     // 64 x 512 f16
    _Float16* H2s = (_Float16*)(smem + 80 * 1024);     // 64 x 512 f16
    float*    Es  = (float*)smem;                      // 64 x 128 f32 (alias)

    const int e0   = blockIdx.x * TM;                  // EE % 64 == 0
    const int tid  = threadIdx.x;
    const int wave = tid >> 5;
    const int lane = tid & 31;

    // stage X = [dist_emb | u[row]-u[col]]  (64 x 128, f16)
    for (int i = tid; i < TM * DIM; i += 256) {
        int m = i >> 7, f = i & 127, e = e0 + m;
        float v;
        if (f < R_DIM) v = dist[(size_t)e * R_DIM + f];
        else {
            int r = ei[e], c = ei[EE + e];
            v = u[(size_t)r * U_DIM + (f - R_DIM)] - u[(size_t)c * U_DIM + (f - R_DIM)];
        }
        Xs[i] = (_Float16)v;
    }
    __syncthreads();

    // layer 1: [64x128] @ [128x512] -> H1   (K tiles = 4)
#pragma unroll
    for (int th = 0; th < 2; ++th) {
        v8f acc[2][4] = {};
#pragma unroll
        for (int tk = 0; tk < 4; ++tk) {
            v16h a[4];
#pragma unroll
            for (int mb = 0; mb < 4; ++mb) a[mb] = load_a_lds(Xs, DIM, mb * 16, tk * 32);
#pragma unroll
            for (int t = 0; t < 2; ++t) {
                v16h bfr = load_b(W1, 4, tk, wave * 4 + th * 2 + t);
#pragma unroll
                for (int mb = 0; mb < 4; ++mb) acc[t][mb] = wmma_f16(a[mb], bfr, acc[t][mb]);
            }
        }
#pragma unroll
        for (int t = 0; t < 2; ++t)
#pragma unroll
            for (int mb = 0; mb < 4; ++mb)
                store_d_relu(H1s, H_E, mb * 16, (wave * 4 + th * 2 + t) * 16, acc[t][mb], b1);
    }
    __syncthreads();

    // layer 2: [64x512] @ [512x512] -> H2   (K tiles = 16)
#pragma unroll
    for (int th = 0; th < 2; ++th) {
        v8f acc[2][4] = {};
#pragma unroll
        for (int tk = 0; tk < 16; ++tk) {
            if (tk + 1 < 16)   // prefetch next k-tile of this half's W2 slice
                __builtin_prefetch(W2 + ((size_t)((wave * 4 + th * 2) * 16 + tk + 1) * 512), 0, 0);
            v16h a[4];
#pragma unroll
            for (int mb = 0; mb < 4; ++mb) a[mb] = load_a_lds(H1s, H_E, mb * 16, tk * 32);
#pragma unroll
            for (int t = 0; t < 2; ++t) {
                v16h bfr = load_b(W2, 16, tk, wave * 4 + th * 2 + t);
#pragma unroll
                for (int mb = 0; mb < 4; ++mb) acc[t][mb] = wmma_f16(a[mb], bfr, acc[t][mb]);
            }
        }
#pragma unroll
        for (int t = 0; t < 2; ++t)
#pragma unroll
            for (int mb = 0; mb < 4; ++mb)
                store_d_relu(H2s, H_E, mb * 16, (wave * 4 + th * 2 + t) * 16, acc[t][mb], b2);
    }
    __syncthreads();

    // layer 3: [64x512] @ [512x128] -> edge_emb  (one 16-col tile per wave, 4 M-blocks)
    {
        v8f acc[4] = {};
#pragma unroll
        for (int tk = 0; tk < 16; ++tk) {
            v16h bfr = load_b(W3, 16, tk, wave);
#pragma unroll
            for (int mb = 0; mb < 4; ++mb)
                acc[mb] = wmma_f16(load_a_lds(H2s, H_E, mb * 16, tk * 32), bfr, acc[mb]);
        }
        int n  = wave * 16 + (lane & 15);
        int mh = (lane < 16) ? 0 : 8;
        float bv = b3[n];
#pragma unroll
        for (int mb = 0; mb < 4; ++mb) {
#pragma unroll
            for (int i = 0; i < 8; ++i) {
                float v = acc[mb][i] + bv;
                Es[(mb * 16 + mh + i) * OUT_E + n] = v;
                embH[(size_t)(e0 + mb * 16 + mh + i) * OUT_E + n] = (_Float16)v;
            }
        }
    }
    __syncthreads();

    // LayerNorm + leaky-relu + alpha projection; each wave handles 8 edges
    for (int s = 0; s < 8; ++s) {
        int m = wave * 8 + s;
        const float* x = Es + m * OUT_E;
        float s1 = 0.f, s2 = 0.f;
        for (int f = lane; f < OUT_E; f += 32) { float v = x[f]; s1 += v; s2 += v * v; }
#pragma unroll
        for (int off = 16; off; off >>= 1) { s1 += __shfl_xor(s1, off, 32); s2 += __shfl_xor(s2, off, 32); }
        float mu  = s1 * (1.f / OUT_E);
        float var = s2 * (1.f / OUT_E) - mu * mu;
        float rs  = rsqrtf(var + 1e-5f);
        float dot = 0.f;
        for (int f = lane; f < OUT_E; f += 32) {
            float v = (x[f] - mu) * rs * ln_g[f] + ln_b[f];
            v = v > 0.f ? v : 0.01f * v;
            dot += v * aW[f];
        }
#pragma unroll
        for (int off = 16; off; off >>= 1) dot += __shfl_xor(dot, off, 32);
        if (lane == 0) {
            float a = dot + ab[0];
            int e = e0 + m;
            logit[e] = a;
            atomicMaxF(nodeMax + ei[EE + e], a);
        }
    }
}

// --------------------------- node MLP (WMMA, fused) ------------------------
__global__ __launch_bounds__(256) void node_mlp(
    const float* __restrict__ u,
    const _Float16* __restrict__ W1, const float* __restrict__ b1,
    const _Float16* __restrict__ W2, const float* __restrict__ b2,
    const _Float16* __restrict__ W3, const float* __restrict__ b3,
    float* __restrict__ out_node)
{
    __shared__ __align__(16) _Float16 Us [16 * U_DIM]; // 2 KB
    __shared__ __align__(16) _Float16 H1s[16 * H_N];   // 8 KB
    __shared__ __align__(16) _Float16 H2s[16 * H_N];   // 8 KB

    const int n0   = blockIdx.x * 16;
    const int tid  = threadIdx.x;
    const int wave = tid >> 5;
    const int lane = tid & 31;

    for (int i = tid; i < 16 * U_DIM; i += 256) {
        int m = i >> 6, f = i & 63, node = n0 + m;
        Us[i] = (_Float16)(node < NN ? u[(size_t)node * U_DIM + f] : 0.f);
    }
    __syncthreads();

    // layer 1: K tiles = 2, 16 N-tiles -> 2 per wave
#pragma unroll
    for (int t = 0; t < 2; ++t) {
        int tn = wave * 2 + t;
        v8f acc = {};
#pragma unroll
        for (int tk = 0; tk < 2; ++tk)
            acc = wmma_f16(load_a_lds(Us, U_DIM, 0, tk * 32), load_b(W1, 2, tk, tn), acc);
        store_d_relu(H1s, H_N, 0, tn * 16, acc, b1);
    }
    __syncthreads();

    // layer 2: K tiles = 8
#pragma unroll
    for (int t = 0; t < 2; ++t) {
        int tn = wave * 2 + t;
        v8f acc = {};
#pragma unroll
        for (int tk = 0; tk < 8; ++tk)
            acc = wmma_f16(load_a_lds(H1s, H_N, 0, tk * 32), load_b(W2, 8, tk, tn), acc);
        store_d_relu(H2s, H_N, 0, tn * 16, acc, b2);
    }
    __syncthreads();

    // layer 3: 4 N-tiles (64 cols), waves 0-3 (wave-uniform branch: EXEC stays all-1)
    if (wave < 4) {
        v8f acc = {};
#pragma unroll
        for (int tk = 0; tk < 8; ++tk)
            acc = wmma_f16(load_a_lds(H2s, H_N, 0, tk * 32), load_b(W3, 8, tk, wave), acc);
        int n  = wave * 16 + (lane & 15);
        int mb = (lane < 16) ? 0 : 8;
        float bv = b3[n];
#pragma unroll
        for (int i = 0; i < 8; ++i) {
            int node = n0 + mb + i;
            if (node < NN) out_node[(size_t)node * N_SC + n] = acc[i] + bv;
        }
    }
}

// ------------------------- segment softmax: exp pass -----------------------
__global__ void softmax_ex(const float* __restrict__ logit, const int* __restrict__ ei,
                           const float* __restrict__ nodeMax,
                           float* __restrict__ exBuf, float* __restrict__ nodeSum) {
    int e = blockIdx.x * 256 + threadIdx.x;
    if (e >= EE) return;
    int c = ei[EE + e];
    float x = expf(logit[e] - nodeMax[c]);
    exBuf[e] = x;
    atomicAdd(nodeSum + c, x);
}

// --------------------- alpha-scale + rotate + segment-sum ------------------
// one thread per (edge, rep): 8 outputs each
__global__ void scatter_rot(const _Float16* __restrict__ embH, const float* __restrict__ exBuf,
                            const float* __restrict__ nodeSum, const float* __restrict__ rot,
                            const int* __restrict__ ei, float* __restrict__ out_edge) {
    int idx = blockIdx.x * 256 + threadIdx.x;
    if (idx >= EE * 16) return;
    int e = idx >> 4, j = idx & 15;
    int c = ei[EE + e];
    float alpha = exBuf[e] / (nodeSum[c] + 1e-16f);
    const _Float16* em = embH + (size_t)e * OUT_E + j * 8;
    const float*    R  = rot  + (size_t)e * 16;          // [4][2][2]
    float*          o  = out_edge + (size_t)c * OUT_E + j * 8;
#pragma unroll
    for (int k = 0; k < 4; ++k) {
        float x = (float)em[k * 2 + 0] * alpha;
        float y = (float)em[k * 2 + 1] * alpha;
        float r00 = R[k*4+0], r01 = R[k*4+1], r10 = R[k*4+2], r11 = R[k*4+3];
        atomicAdd(o + k * 2 + 0, x * r00 + y * r01);     // rot_inv = rot^T
        atomicAdd(o + k * 2 + 1, x * r10 + y * r11);
    }
}

// --------------------------------- launcher --------------------------------
extern "C" void kernel_launch(void* const* d_in, const int* in_sizes, int n_in,
                              void* d_out, int out_size, void* d_ws, size_t ws_size,
                              hipStream_t stream) {
    const float* dist = (const float*)d_in[0];
    const float* u    = (const float*)d_in[1];
    const float* rot  = (const float*)d_in[2];
    const int*   ei   = (const int*)d_in[3];
    const float* eW1 = (const float*)d_in[4];  const float* eb1 = (const float*)d_in[5];
    const float* eW2 = (const float*)d_in[6];  const float* eb2 = (const float*)d_in[7];
    const float* eW3 = (const float*)d_in[8];  const float* eb3 = (const float*)d_in[9];
    const float* nW1 = (const float*)d_in[10]; const float* nb1 = (const float*)d_in[11];
    const float* nW2 = (const float*)d_in[12]; const float* nb2 = (const float*)d_in[13];
    const float* nW3 = (const float*)d_in[14]; const float* nb3 = (const float*)d_in[15];
    const float* ln_g = (const float*)d_in[16]; const float* ln_b = (const float*)d_in[17];
    const float* aW   = (const float*)d_in[18]; const float* ab   = (const float*)d_in[19];

    float* out_node = (float*)d_out;                     // [N, 64]
    float* out_edge = (float*)d_out + (size_t)NN * N_SC; // [N, 128]

    // workspace carve-up (~107 MB)
    char* ws = (char*)d_ws; size_t o = 0;
    _Float16* eW1h = (_Float16*)(ws + o); o += (size_t)DIM * H_E * 2;
    _Float16* eW2h = (_Float16*)(ws + o); o += (size_t)H_E * H_E * 2;
    _Float16* eW3h = (_Float16*)(ws + o); o += (size_t)H_E * OUT_E * 2;
    _Float16* nW1h = (_Float16*)(ws + o); o += (size_t)U_DIM * H_N * 2;
    _Float16* nW2h = (_Float16*)(ws + o); o += (size_t)H_N * H_N * 2;
    _Float16* nW3h = (_Float16*)(ws + o); o += (size_t)H_N * N_SC * 2;
    _Float16* embH = (_Float16*)(ws + o); o += (size_t)EE * OUT_E * 2;
    float* logit   = (float*)(ws + o);    o += (size_t)EE * 4;
    float* exBuf   = (float*)(ws + o);    o += (size_t)EE * 4;
    float* nodeMax = (float*)(ws + o);    o += (size_t)NN * 4;
    float* nodeSum = (float*)(ws + o);    o += (size_t)NN * 4;

    // 1) pre-pack weights into B-fragment order
    prep_weight<<<(DIM*H_E + 255)/256, 256, 0, stream>>>(eW1, eW1h, DIM, H_E);
    prep_weight<<<(H_E*H_E + 255)/256, 256, 0, stream>>>(eW2, eW2h, H_E, H_E);
    prep_weight<<<(H_E*OUT_E + 255)/256, 256, 0, stream>>>(eW3, eW3h, H_E, OUT_E);
    prep_weight<<<(U_DIM*H_N + 255)/256, 256, 0, stream>>>(nW1, nW1h, U_DIM, H_N);
    prep_weight<<<(H_N*H_N + 255)/256, 256, 0, stream>>>(nW2, nW2h, H_N, H_N);
    prep_weight<<<(H_N*N_SC + 255)/256, 256, 0, stream>>>(nW3, nW3h, H_N, N_SC);

    // 2) init reduction buffers + zero out_edge (replay-safe)
    init_kernel<<<(NN*OUT_E + 255)/256, 256, 0, stream>>>(nodeMax, nodeSum, out_edge);

    // 3) fused edge MLP + LN + alpha logits + segment max
    edge_mlp<<<EE/TM, 256, 0, stream>>>(dist, u, ei, eW1h, eb1, eW2h, eb2, eW3h, eb3,
                                        ln_g, ln_b, aW, ab, embH, logit, nodeMax);

    // 4) node MLP (independent)
    node_mlp<<<(NN + 15)/16, 256, 0, stream>>>(u, nW1h, nb1, nW2h, nb2, nW3h, nb3, out_node);

    // 5) softmax exp + segment sum
    softmax_ex<<<(EE + 255)/256, 256, 0, stream>>>(logit, ei, nodeMax, exBuf, nodeSum);

    // 6) alpha-normalize, rotate, segment-sum scatter
    scatter_rot<<<(EE*16 + 255)/256, 256, 0, stream>>>(embH, exBuf, nodeSum, rot, ei, out_edge);
}
